// SNNPolicy_73014444032534
// MI455X (gfx1250) — matrix operand order repaired and checked by reading, not compiled, not run on hardware
//
#include <hip/hip_runtime.h>

// ---------------------------------------------------------------------------
// SNN policy forward, fused for MI455X (gfx1250, wave32, WMMA bf16).
// B=32768, OBS=512, H1=H2=128, A=18, T=30.
//
// Roofline: mandatory HBM traffic = obs (64 MB) + out (2.25 MB) ~ 2.8 us at
// 23.3 TB/s; 41 GFLOP of bf16 WMMA is far below the matrix ceiling, so the
// kernel streams obs exactly once, holds W2/Wout in LDS, and keeps h1/mem1/
// mem2/acc in VGPRs across all 30 timesteps. mem2 is pre-decayed elementwise
// and used as the WMMA C operand so no separate h2 accumulator exists.
// ---------------------------------------------------------------------------

#define OBS_D   512
#define H1_D    128
#define H2_D    128
#define NA_D    18
#define NAP_D   32          // Wout rows padded to 32 (two 16-col WMMA tiles)
#define TSTEPS  30
#define BETA_C  0.9f
#define LPITCH  136         // LDS row pitch (elements) to spread banks

typedef __attribute__((ext_vector_type(16))) __bf16 v16bf;
typedef __attribute__((ext_vector_type(8)))  __bf16 v8bf;
typedef __attribute__((ext_vector_type(8)))  float  v8f;

static __device__ __forceinline__ __bf16 f2bf(float f) {
    return (__bf16)f;   // native HW convert (v_cvt_pk_bf16_f32 when paired)
}

static __device__ __forceinline__ v16bf load_bf16x16(const __bf16* p) {
    // two 16B loads (16B-aligned by construction; LDS pitch 136*2B = 272B = 17*16B)
    v8bf lo = *(const v8bf*)p;
    v8bf hi = *(const v8bf*)(p + 16);
    v16bf r;
#pragma unroll
    for (int e = 0; e < 8; ++e) { r[e] = lo[e]; r[8 + e] = hi[e]; }
    return r;
}

// Build a 16x32 bf16 A-fragment from a wave-private LDS slab S[16][LPITCH].
// ISA A-layout (16-bit, 16x32): lane L<16 -> row M=L, K = [k0+8h, +8) and
// [k0+16+8h, +8) where h = lane>>4.
static __device__ __forceinline__ v16bf lds_afrag(const __bf16* Sw, int l16,
                                                  int half, int k0) {
    const __bf16* p0 = Sw + l16 * LPITCH + k0 + half * 8;
    const __bf16* p1 = Sw + l16 * LPITCH + k0 + 16 + half * 8;
    v8bf lo = *(const v8bf*)p0;
    v8bf hi = *(const v8bf*)p1;
    v16bf r;
#pragma unroll
    for (int e = 0; e < 8; ++e) { r[e] = lo[e]; r[8 + e] = hi[e]; }
    return r;
}

// ---------------------------------------------------------------------------
// Prep: convert W1 (f32, 128x512) -> bf16 in workspace.
// ---------------------------------------------------------------------------
__global__ void cvt_w1_bf16(const float* __restrict__ w, __bf16* __restrict__ o,
                            int n) {
    int i = blockIdx.x * blockDim.x + threadIdx.x;
    if (i < n) o[i] = f2bf(w[i]);
}

// ---------------------------------------------------------------------------
// Fused SNN forward. 256 threads = 8 waves; each wave owns 16 batch rows,
// block owns 128 rows. grid.x = B/128.
// ---------------------------------------------------------------------------
__global__ __launch_bounds__(256) void snn_fused(
    const float*  __restrict__ obs,   // [B, 512]
    const __bf16* __restrict__ w1b,   // [128, 512] bf16 (from prep)
    const float*  __restrict__ b1,    // [128]
    const float*  __restrict__ W2,    // [128, 128]
    const float*  __restrict__ b2,    // [128]
    const float*  __restrict__ Wout,  // [18, 128]
    const float*  __restrict__ bout,  // [18]
    float*        __restrict__ out)   // [B, 18]
{
    __shared__ __align__(16) __bf16 W2s[H2_D * LPITCH];       // 34.0 KB
    __shared__ __align__(16) __bf16 Wos[NAP_D * LPITCH];      //  8.5 KB
    __shared__ __align__(16) __bf16 Ss[8 * 16 * LPITCH];      // 34.0 KB (per-wave slabs)

    const int tid = threadIdx.x;

    // Stage W2 (bf16) and zero-padded Wout rows into LDS.
    for (int i = tid; i < H2_D * H1_D; i += 256) {
        int r = i >> 7, c = i & 127;
        W2s[r * LPITCH + c] = f2bf(W2[i]);
    }
    for (int i = tid; i < NAP_D * H2_D; i += 256) {
        int r = i >> 7, c = i & 127;
        float v = (r < NA_D) ? Wout[r * H2_D + c] : 0.0f;
        Wos[r * LPITCH + c] = f2bf(v);
    }
    __syncthreads();

    const int lane = tid & 31;
    const int wave = tid >> 5;
    const int half = lane >> 4;   // K-half / M-half selector
    const int l16  = lane & 15;
    const size_t rowBase = (size_t)blockIdx.x * 128 + (size_t)wave * 16;

    __bf16* Sw = Ss + wave * 16 * LPITCH;  // wave-private shuffle slab

    // ---------------- Phase A: h1 = obs @ W1^T + b1 (bf16 WMMA) -------------
    // Fold b1 into the initial accumulators (column = j*16+l16 everywhere).
    v8f h1a[8];
#pragma unroll
    for (int j = 0; j < 8; ++j) {
        const float bb = b1[j * 16 + l16];
#pragma unroll
        for (int e = 0; e < 8; ++e) h1a[j][e] = bb;
    }

    const float* arow = obs + (rowBase + (size_t)l16) * OBS_D;
    for (int k0 = 0; k0 < OBS_D; k0 += 32) {
        const float* ap = arow + k0 + half * 8;
        __builtin_prefetch((const void*)(ap + 32), 0, 3);  // global_prefetch_b8
        const float4 p0 = *(const float4*)(ap + 0);
        const float4 p1 = *(const float4*)(ap + 4);
        const float4 p2 = *(const float4*)(ap + 16);
        const float4 p3 = *(const float4*)(ap + 20);
        v16bf af;
        af[0]  = f2bf(p0.x); af[1]  = f2bf(p0.y); af[2]  = f2bf(p0.z); af[3]  = f2bf(p0.w);
        af[4]  = f2bf(p1.x); af[5]  = f2bf(p1.y); af[6]  = f2bf(p1.z); af[7]  = f2bf(p1.w);
        af[8]  = f2bf(p2.x); af[9]  = f2bf(p2.y); af[10] = f2bf(p2.z); af[11] = f2bf(p2.w);
        af[12] = f2bf(p3.x); af[13] = f2bf(p3.y); af[14] = f2bf(p3.z); af[15] = f2bf(p3.w);
#pragma unroll
        for (int j = 0; j < 8; ++j) {
            // B layout: lane = column N (= W1 row j*16+l16), K = [k0+8h,+8) and [k0+16+8h,+8)
            const __bf16* bp = w1b + (size_t)(j * 16 + l16) * OBS_D + k0 + half * 8;
            const v16bf bf = load_bf16x16(bp);
            h1a[j] = __builtin_amdgcn_wmma_f32_16x16x32_bf16(
                false, af, false, bf, (short)0, h1a[j], false, false);
        }
    }

    // ---------------- Phase B: 30 leaky-integrate-and-fire steps ------------
    v8f m1[8], m2[8], acq[2];
#pragma unroll
    for (int j = 0; j < 8; ++j)
#pragma unroll
        for (int e = 0; e < 8; ++e) { m1[j][e] = 0.0f; m2[j][e] = 0.0f; }
#pragma unroll
    for (int j = 0; j < 2; ++j)
#pragma unroll
        for (int e = 0; e < 8; ++e) acq[j][e] = 0.0f;

    float b2v[8];
#pragma unroll
    for (int j = 0; j < 8; ++j) b2v[j] = b2[j * 16 + l16];

    const __bf16 one_bf  = (__bf16)1.0f;
    const __bf16 zero_bf = (__bf16)0.0f;

    for (int t = 0; t < TSTEPS; ++t) {
        // mem1 update + spike, scatter spikes into A-layout staging (LDS)
#pragma unroll
        for (int j = 0; j < 8; ++j) {
#pragma unroll
            for (int e = 0; e < 8; ++e) {
                float m = m1[j][e];
                float r = (m > 1.0f) ? 1.0f : 0.0f;     // detached reset
                m = BETA_C * m + h1a[j][e] - r;
                m1[j][e] = m;
                // spike(mem-thr>0): exact in bf16, just select constants
                Sw[(e + half * 8) * LPITCH + j * 16 + l16] =
                    (m > 1.0f) ? one_bf : zero_bf;
            }
        }
        // Pre-decay mem2 in place: mem2 <- beta*mem2 - r2 + b2.
        // The GEMM below then accumulates spk1@W2^T directly into mem2
        // (WMMA C operand), so no separate h2 register tile is needed.
#pragma unroll
        for (int j = 0; j < 8; ++j) {
#pragma unroll
            for (int e = 0; e < 8; ++e) {
                float m = m2[j][e];
                float r = (m > 1.0f) ? 1.0f : 0.0f;
                m2[j][e] = BETA_C * m - r + b2v[j];
            }
        }
#pragma unroll
        for (int kc = 0; kc < 4; ++kc) {
            const int k0 = kc * 32;
            const v16bf af = lds_afrag(Sw, l16, half, k0);
#pragma unroll
            for (int j = 0; j < 8; ++j) {
                const v16bf bf =
                    load_bf16x16(W2s + (j * 16 + l16) * LPITCH + k0 + half * 8);
                m2[j] = __builtin_amdgcn_wmma_f32_16x16x32_bf16(
                    false, af, false, bf, (short)0, m2[j], false, false);
            }
        }
        // Stage mem2 (bf16) for the q GEMM in the same wave-private slab.
#pragma unroll
        for (int j = 0; j < 8; ++j) {
#pragma unroll
            for (int e = 0; e < 8; ++e) {
                Sw[(e + half * 8) * LPITCH + j * 16 + l16] = f2bf(m2[j][e]);
            }
        }
        // acc += mem2 @ Wout^T   (bout folded in at the end)
#pragma unroll
        for (int kc = 0; kc < 4; ++kc) {
            const int k0 = kc * 32;
            const v16bf af = lds_afrag(Sw, l16, half, k0);
#pragma unroll
            for (int j = 0; j < 2; ++j) {
                const v16bf bf =
                    load_bf16x16(Wos + (j * 16 + l16) * LPITCH + k0 + half * 8);
                acq[j] = __builtin_amdgcn_wmma_f32_16x16x32_bf16(
                    false, af, false, bf, (short)0, acq[j], false, false);
            }
        }
    }

    // ---------------- Epilogue: out = acc/T + bout --------------------------
    const float inv = 1.0f / (float)TSTEPS;
#pragma unroll
    for (int j = 0; j < 2; ++j) {
        const int n = j * 16 + l16;
        if (n < NA_D) {
            const float bo = bout[n];
#pragma unroll
            for (int e = 0; e < 8; ++e) {
                const size_t row = rowBase + (size_t)(e + half * 8);
                out[row * NA_D + n] = acq[j][e] * inv + bo;
            }
        }
    }
}

// ---------------------------------------------------------------------------
extern "C" void kernel_launch(void* const* d_in, const int* in_sizes, int n_in,
                              void* d_out, int out_size, void* d_ws, size_t ws_size,
                              hipStream_t stream) {
    const float* obs  = (const float*)d_in[0];
    const float* W1   = (const float*)d_in[1];
    const float* b1   = (const float*)d_in[2];
    const float* W2   = (const float*)d_in[3];
    const float* b2   = (const float*)d_in[4];
    const float* Wout = (const float*)d_in[5];
    const float* bout = (const float*)d_in[6];
    float* out = (float*)d_out;

    __bf16* w1b = (__bf16*)d_ws;                 // 128*512 bf16 = 128 KB scratch

    const int n_w1 = H1_D * OBS_D;               // 65536
    cvt_w1_bf16<<<(n_w1 + 255) / 256, 256, 0, stream>>>(W1, w1b, n_w1);

    const int B = in_sizes[0] / OBS_D;           // 32768
    const int grid = B / 128;                    // 256 blocks, 8 waves each
    snn_fused<<<grid, 256, 0, stream>>>(obs, w1b, b1, W2, b2, Wout, bout, out);
}
